// SpaceFormerImputation_1400159338538
// MI455X (gfx1250) — compile-verified, matrix operand order checked by reading, not run on hardware
//
#include <hip/hip_runtime.h>

// ---------------------------------------------------------------------------
// SpaceFormer imputation pipeline for MI455X (gfx1250), bf16 WMMA everywhere.
// N=8192, D=512, F=2048, G=102.
// Round 2: templated epilogues (no runtime-null branches around WMMA stores),
// software-pipelined global->LDS staging, global_prefetch hints.
// ---------------------------------------------------------------------------

#define NTOK 8192
#define DDIM 512
#define FDIM 2048
#define GDIM 102
#define GAMMA_C 0.5f
#define NEGV -1e9f
#define SCALE_C 0.044194173824159216f  // 1/sqrt(512)

typedef __attribute__((ext_vector_type(16))) __bf16 v16bf;
typedef __attribute__((ext_vector_type(8)))  float  v8f;

#define WMMA_BF16(a, b, c) \
    __builtin_amdgcn_wmma_f32_16x16x32_bf16(false, (a), false, (b), (short)0, (c), false, false)

// ---- monotone float<->uint mapping for atomicMax on floats --------------
__device__ __forceinline__ unsigned f2mono(float f) {
    unsigned u = __float_as_uint(f);
    return (u >> 31) ? ~u : (u | 0x80000000u);
}
__device__ __forceinline__ float mono2f(unsigned u) {
    return (u >> 31) ? __uint_as_float(u & 0x7fffffffu) : __uint_as_float(~u);
}

// ---- fragment load from LDS, per ISA 7.12.2 16-bit A/B layout ------------
// lane (l&15) selects the matrix row (A) / column (B); caller passes that in
// `row`. VGPR vg holds k-pair: k = (vg&3)*2 + (vg>>2)*16 + (lane>>4)*8.
// k offsets per half: {0,2,4,6}+{16,18,20,22} -> two contiguous 16B groups,
// which the compiler merges into 2x ds_load_b128 (verified in round-1 asm).
__device__ __forceinline__ v16bf load_frag(const __bf16* base, int row, int stride) {
    int hi = (threadIdx.x & 31) >> 4;
    union { v16bf v; unsigned u[8]; } f;
    const __bf16* p = base + row * stride;
#pragma unroll
    for (int vg = 0; vg < 8; ++vg) {
        int k = ((vg & 3) * 2) + ((vg >> 2) * 16) + hi * 8;
        f.u[vg] = *(const unsigned*)(p + k);
    }
    return f.v;
}

// ===========================================================================
// Generic bf16 WMMA GEMM:  C[M,Nc] = epi( A[M,K] @ Bt[Nc,K]^T + bias )
// Block tile 128x128, 8 waves, wave tile 32x64 (2x4 16x16 frags).
// M % 128 == 0, Nc % 128 == 0, K % 32 == 0 (true for all calls here).
// EPI bits: 1=bias  2=gelu  4=store f32  8=store bf16  16=store bf16^T
// ===========================================================================
template <int EPI>
__global__ __launch_bounds__(256) void gemm_bf16_kernel(
    const __bf16* __restrict__ A, const __bf16* __restrict__ Bt,
    const float* __restrict__ bias,
    float* __restrict__ outF, __bf16* __restrict__ outB, __bf16* __restrict__ outBT,
    int M, int Nc, int K)
{
    __shared__ __bf16 sA[128 * 40];
    __shared__ __bf16 sB[128 * 40];

    const int t = threadIdx.x;
    const int rowBlk = blockIdx.x * 128;
    const int colBlk = blockIdx.y * 128;
    const int wave = t >> 5, lane = t & 31;
    const int wm = wave & 3, wn = wave >> 2;      // 4 x 2 wave grid
    const int ln = lane & 15, hi = lane >> 4;

    v8f acc[2][4] = {};

    const int lrow = t >> 1, half = t & 1;        // 128 rows x 2 halves of 16

    // per-thread staging pointers (element offset k0 advances the tile)
    const __bf16* pa = A  + (size_t)(rowBlk + lrow) * K + half * 16;
    const __bf16* pb = Bt + (size_t)(colBlk + lrow) * K + half * 16;
    uint4* qa = (uint4*)&sA[lrow * 40 + half * 16];
    uint4* qb = (uint4*)&sB[lrow * 40 + half * 16];

    // prologue: stage first K-tile into registers
    uint4 ra0 = ((const uint4*)pa)[0], ra1 = ((const uint4*)pa)[1];
    uint4 rb0 = ((const uint4*)pb)[0], rb1 = ((const uint4*)pb)[1];

    for (int k0 = 0; k0 < K; k0 += 32) {
        qa[0] = ra0; qa[1] = ra1;
        qb[0] = rb0; qb[1] = rb1;
        __syncthreads();

        // issue next tile's global loads now; they retire while we WMMA
        if (k0 + 32 < K) {
            const uint4* na = (const uint4*)(pa + k0 + 32);
            const uint4* nb = (const uint4*)(pb + k0 + 32);
            ra0 = na[0]; ra1 = na[1];
            rb0 = nb[0]; rb1 = nb[1];
            if (k0 + 64 < K) {
                __builtin_prefetch(pa + k0 + 64, 0, 1);   // global_prefetch_b8
                __builtin_prefetch(pb + k0 + 64, 0, 1);
            }
        }

        v16bf fA[2], fB[4];
#pragma unroll
        for (int mi = 0; mi < 2; ++mi) fA[mi] = load_frag(sA, wm * 32 + mi * 16 + ln, 40);
#pragma unroll
        for (int ni = 0; ni < 4; ++ni) fB[ni] = load_frag(sB, wn * 64 + ni * 16 + ln, 40);
#pragma unroll
        for (int mi = 0; mi < 2; ++mi)
#pragma unroll
            for (int ni = 0; ni < 4; ++ni)
                acc[mi][ni] = WMMA_BF16(fA[mi], fB[ni], acc[mi][ni]);
        __syncthreads();
    }

    // epilogue: C frag layout: reg r -> row = r + 8*hi, col = ln
#pragma unroll
    for (int mi = 0; mi < 2; ++mi) {
#pragma unroll
        for (int ni = 0; ni < 4; ++ni) {
            int col = colBlk + wn * 64 + ni * 16 + ln;
            float bv = 0.0f;
            if constexpr (EPI & 1) bv = bias[col];
#pragma unroll
            for (int r = 0; r < 8; ++r) {
                int row = rowBlk + wm * 32 + mi * 16 + r + 8 * hi;
                float v = acc[mi][ni][r] + bv;
                if constexpr (EPI & 2) v = 0.5f * v * (1.0f + erff(v * 0.70710678118654752f));
                if constexpr (EPI & 4)  outF[(size_t)row * Nc + col] = v;
                if constexpr (EPI & 8)  outB[(size_t)row * Nc + col] = (__bf16)v;
                if constexpr (EPI & 16) outBT[(size_t)col * M + row] = (__bf16)v;
            }
        }
    }
}

// ===========================================================================
// Dual-mask attention score kernel. Block tile 128(i) x 64(j), 8 waves in a
// 4x2 grid, wave tile 32x32 (2x2 frags), two accumulator sets (real/fake).
// MODE 0: atomicMax row maxima of masked scores (both matrices).
// MODE 1: P[i,j] = exp(sr-m) + GAMMA*exp(sf-m) (bf16) + atomicAdd row sums.
// ===========================================================================
template <int MODE>
__global__ __launch_bounds__(256) void attn_scores_kernel(
    const __bf16* __restrict__ Qr, const __bf16* __restrict__ Kr,
    const __bf16* __restrict__ Qf, const __bf16* __restrict__ Kf,
    const unsigned char* __restrict__ rmask, const unsigned char* __restrict__ fmask,
    unsigned* __restrict__ rowmax_u, float* __restrict__ rowsum,
    __bf16* __restrict__ P)
{
    __shared__ __bf16 sQr[128 * 40];
    __shared__ __bf16 sQf[128 * 40];
    __shared__ __bf16 sKr[64 * 40];
    __shared__ __bf16 sKf[64 * 40];

    const int t = threadIdx.x;
    const int iBlk = blockIdx.x * 128;   // query rows
    const int jBlk = blockIdx.y * 64;    // key cols
    const int wave = t >> 5, lane = t & 31;
    const int wm = wave & 3, wn = wave >> 2;
    const int ln = lane & 15, hi = lane >> 4;

    v8f aR[2][2] = {};
    v8f aF[2][2] = {};

    const int qrow = t >> 1, qhalf = t & 1;   // 128 rows x 2 halves
    const int krow = t >> 2, kq = t & 3;      // 64 rows x 4 quarters

    const __bf16* pQr = Qr + (size_t)(iBlk + qrow) * DDIM + qhalf * 16;
    const __bf16* pQf = Qf + (size_t)(iBlk + qrow) * DDIM + qhalf * 16;
    const __bf16* pKr = Kr + (size_t)(jBlk + krow) * DDIM + kq * 8;
    const __bf16* pKf = Kf + (size_t)(jBlk + krow) * DDIM + kq * 8;
    uint4* qQr = (uint4*)&sQr[qrow * 40 + qhalf * 16];
    uint4* qQf = (uint4*)&sQf[qrow * 40 + qhalf * 16];
    uint4* qKr = (uint4*)&sKr[krow * 40 + kq * 8];
    uint4* qKf = (uint4*)&sKf[krow * 40 + kq * 8];

    // prologue staging
    uint4 rq0 = ((const uint4*)pQr)[0], rq1 = ((const uint4*)pQr)[1];
    uint4 rf0 = ((const uint4*)pQf)[0], rf1 = ((const uint4*)pQf)[1];
    uint4 rk0 = ((const uint4*)pKr)[0];
    uint4 rk1 = ((const uint4*)pKf)[0];

    for (int k0 = 0; k0 < DDIM; k0 += 32) {
        qQr[0] = rq0; qQr[1] = rq1;
        qQf[0] = rf0; qQf[1] = rf1;
        qKr[0] = rk0;
        qKf[0] = rk1;
        __syncthreads();

        if (k0 + 32 < DDIM) {
            rq0 = ((const uint4*)(pQr + k0 + 32))[0];
            rq1 = ((const uint4*)(pQr + k0 + 32))[1];
            rf0 = ((const uint4*)(pQf + k0 + 32))[0];
            rf1 = ((const uint4*)(pQf + k0 + 32))[1];
            rk0 = ((const uint4*)(pKr + k0 + 32))[0];
            rk1 = ((const uint4*)(pKf + k0 + 32))[0];
            if (k0 + 64 < DDIM) {
                __builtin_prefetch(pQr + k0 + 64, 0, 1);
                __builtin_prefetch(pKr + k0 + 64, 0, 1);
            }
        }

        v16bf fQr[2], fQf[2], fKr[2], fKf[2];
#pragma unroll
        for (int mi = 0; mi < 2; ++mi) {
            fQr[mi] = load_frag(sQr, wm * 32 + mi * 16 + ln, 40);
            fQf[mi] = load_frag(sQf, wm * 32 + mi * 16 + ln, 40);
        }
#pragma unroll
        for (int ni = 0; ni < 2; ++ni) {
            fKr[ni] = load_frag(sKr, wn * 32 + ni * 16 + ln, 40);
            fKf[ni] = load_frag(sKf, wn * 32 + ni * 16 + ln, 40);
        }
#pragma unroll
        for (int mi = 0; mi < 2; ++mi)
#pragma unroll
            for (int ni = 0; ni < 2; ++ni) {
                aR[mi][ni] = WMMA_BF16(fQr[mi], fKr[ni], aR[mi][ni]);
                aF[mi][ni] = WMMA_BF16(fQf[mi], fKf[ni], aF[mi][ni]);
            }
        __syncthreads();
    }

    if constexpr (MODE == 0) {
#pragma unroll
        for (int mi = 0; mi < 2; ++mi) {
#pragma unroll
            for (int r = 0; r < 8; ++r) {
                int row = iBlk + wm * 32 + mi * 16 + r + 8 * hi;
                float mx = -3.4e38f;
#pragma unroll
                for (int ni = 0; ni < 2; ++ni) {
                    int col = jBlk + wn * 32 + ni * 16 + ln;
                    size_t midx = (size_t)row * NTOK + col;
                    float sr = rmask[midx] ? NEGV : aR[mi][ni][r] * SCALE_C;
                    float sf = fmask[midx] ? NEGV : aF[mi][ni][r] * SCALE_C;
                    mx = fmaxf(mx, fmaxf(sr, sf));
                }
                for (int s = 1; s < 16; s <<= 1) mx = fmaxf(mx, __shfl_xor(mx, s, 32));
                if (ln == 0) atomicMax(&rowmax_u[row], f2mono(mx));
            }
        }
    } else {
#pragma unroll
        for (int mi = 0; mi < 2; ++mi) {
#pragma unroll
            for (int r = 0; r < 8; ++r) {
                int row = iBlk + wm * 32 + mi * 16 + r + 8 * hi;
                float m = mono2f(rowmax_u[row]);
                float acc = 0.0f;
#pragma unroll
                for (int ni = 0; ni < 2; ++ni) {
                    int col = jBlk + wn * 32 + ni * 16 + ln;
                    size_t midx = (size_t)row * NTOK + col;
                    float sr = rmask[midx] ? NEGV : aR[mi][ni][r] * SCALE_C;
                    float sf = fmask[midx] ? NEGV : aF[mi][ni][r] * SCALE_C;
                    float w = expf(sr - m) + GAMMA_C * expf(sf - m);
                    P[(size_t)row * NTOK + col] = (__bf16)w;
                    acc += w;
                }
                for (int s = 1; s < 16; s <<= 1) acc += __shfl_xor(acc, s, 32);
                if (ln == 0) atomicAdd(&rowsum[row], acc);
            }
        }
    }
}

// ===========================================================================
// Residual + (optional 1/rowsum scale on delta) + LayerNorm. One wave / row.
// ===========================================================================
__global__ __launch_bounds__(256) void ln_kernel(
    const float* __restrict__ base, const float* __restrict__ delta,
    const float* __restrict__ rowsum,
    const float* __restrict__ g, const float* __restrict__ b,
    float* __restrict__ outF, __bf16* __restrict__ outB)
{
    int row = blockIdx.x * 8 + (threadIdx.x >> 5);
    int lane = threadIdx.x & 31;
    float inv = rowsum ? (1.0f / rowsum[row]) : 1.0f;

    float v[16];
    float s = 0.0f;
#pragma unroll
    for (int i = 0; i < 16; ++i) {
        int c = lane + i * 32;
        size_t idx = (size_t)row * DDIM + c;
        v[i] = base[idx] + delta[idx] * inv;
        s += v[i];
    }
    for (int m = 1; m < 32; m <<= 1) s += __shfl_xor(s, m, 32);
    float mu = s * (1.0f / (float)DDIM);
    float s2 = 0.0f;
#pragma unroll
    for (int i = 0; i < 16; ++i) { float d0 = v[i] - mu; s2 += d0 * d0; }
    for (int m = 1; m < 32; m <<= 1) s2 += __shfl_xor(s2, m, 32);
    float rs = rsqrtf(s2 * (1.0f / (float)DDIM) + 1e-5f);
#pragma unroll
    for (int i = 0; i < 16; ++i) {
        int c = lane + i * 32;
        size_t idx = (size_t)row * DDIM + c;
        float o = (v[i] - mu) * rs * g[c] + b[c];
        if (outF) outF[idx] = o;
        if (outB) outB[idx] = (__bf16)o;
    }
}

// ===========================================================================
// Small utility kernels
// ===========================================================================
__global__ void convert_kernel(const float* __restrict__ in,
                               float* __restrict__ outF, __bf16* __restrict__ outB, int n) {
    int i = blockIdx.x * 256 + threadIdx.x;
    if (i >= n) return;
    float v = in[i];
    outF[i] = v;
    outB[i] = (__bf16)v;
}

__global__ void tc_kernel(const float* __restrict__ in, __bf16* __restrict__ out, int R, int C) {
    int i = blockIdx.x * 256 + threadIdx.x;
    if (i >= R * C) return;
    int r = i / C, c = i - r * C;
    out[(size_t)c * R + r] = (__bf16)in[i];  // out is [C,R] = in^T
}

__global__ void zero_drop_kernel(const int* __restrict__ idx,
                                 float* __restrict__ df, __bf16* __restrict__ db, int total) {
    int i = blockIdx.x * 256 + threadIdx.x;
    if (i >= total) return;
    int row = i / GDIM;
    size_t o = (size_t)row * DDIM + idx[i];
    df[o] = 0.0f;
    db[o] = (__bf16)0.0f;
}

__global__ void gather_kernel(const float* __restrict__ src, const int* __restrict__ idx,
                              float* __restrict__ out, int total) {
    int i = blockIdx.x * 256 + threadIdx.x;
    if (i >= total) return;
    int row = i / GDIM;
    out[i] = src[(size_t)row * DDIM + idx[i]];
}

// ===========================================================================
// Host launcher
// ===========================================================================
extern "C" void kernel_launch(void* const* d_in, const int* in_sizes, int n_in,
                              void* d_out, int out_size, void* d_ws, size_t ws_size,
                              hipStream_t stream) {
    const float* x         = (const float*)d_in[0];
    const int*   drop_idx  = (const int*)d_in[1];
    const unsigned char* rmask = (const unsigned char*)d_in[2];
    const unsigned char* fmask = (const unsigned char*)d_in[3];
    const float* enc_W[5]  = { (const float*)d_in[4], (const float*)d_in[5],
                               (const float*)d_in[6], (const float*)d_in[7],
                               (const float*)d_in[8] };   // Wq_r, Wk_r, Wq_f, Wk_f, Wv
    const float* dec_W[5]  = { (const float*)d_in[9], (const float*)d_in[10],
                               (const float*)d_in[11], (const float*)d_in[12],
                               (const float*)d_in[13] };
    const float* ff_W1 = (const float*)d_in[14];
    const float* ff_b1 = (const float*)d_in[15];
    const float* ff_W2 = (const float*)d_in[16];
    const float* ff_b2 = (const float*)d_in[17];
    const float* ln1_g = (const float*)d_in[18];
    const float* ln1_b = (const float*)d_in[19];
    const float* ln2_g = (const float*)d_in[20];
    const float* ln2_b = (const float*)d_in[21];
    const float* ln3_g = (const float*)d_in[22];
    const float* ln3_b = (const float*)d_in[23];
    const float* head_W = (const float*)d_in[24];
    const float* head_b = (const float*)d_in[25];

    const int N = NTOK, D = DDIM, F = FDIM, G = GDIM;
    (void)in_sizes; (void)n_in; (void)out_size; (void)ws_size;

    // ---- workspace bump allocator ----
    char* wsp = (char*)d_ws;
    size_t off = 0;
    auto alloc = [&](size_t bytes) -> void* {
        void* p = wsp + off;
        off = (off + bytes + 255) & ~(size_t)255;
        return p;
    };

    __bf16* encWt[5]; for (int i = 0; i < 5; ++i) encWt[i] = (__bf16*)alloc((size_t)D * D * 2);
    __bf16* decWt[5]; for (int i = 0; i < 5; ++i) decWt[i] = (__bf16*)alloc((size_t)D * D * 2);
    __bf16* headWt = (__bf16*)alloc((size_t)D * D * 2);
    __bf16* W1t = (__bf16*)alloc((size_t)D * F * 2);   // [F,D]
    __bf16* W2t = (__bf16*)alloc((size_t)F * D * 2);   // [D,F]

    float*  dropf  = (float*)alloc((size_t)N * D * 4);
    __bf16* dropb  = (__bf16*)alloc((size_t)N * D * 2);
    __bf16* qr_b   = (__bf16*)alloc((size_t)N * D * 2);
    __bf16* kr_b   = (__bf16*)alloc((size_t)N * D * 2);
    __bf16* qf_b   = (__bf16*)alloc((size_t)N * D * 2);
    __bf16* kf_b   = (__bf16*)alloc((size_t)N * D * 2);
    __bf16* vt_b   = (__bf16*)alloc((size_t)D * N * 2);   // V^T [D,N]
    __bf16* Pmat   = (__bf16*)alloc((size_t)N * N * 2);   // 128 MB
    unsigned* rowmax_u = (unsigned*)alloc((size_t)N * 4);
    float*    rowsum   = (float*)alloc((size_t)N * 4);
    float*  pv_f   = (float*)alloc((size_t)N * D * 4);    // also FF2 out
    float*  hid_f  = (float*)alloc((size_t)N * D * 4);
    __bf16* hid_b  = (__bf16*)alloc((size_t)N * D * 2);
    float*  hid2_f = (float*)alloc((size_t)N * D * 4);
    __bf16* hid2_b = (__bf16*)alloc((size_t)N * D * 2);
    __bf16* ff1_b  = (__bf16*)alloc((size_t)N * F * 2);
    __bf16* recon_b = (__bf16*)alloc((size_t)N * D * 2);
    float*  head_f  = (float*)alloc((size_t)N * D * 4);

    float* out_init  = (float*)d_out;
    float* out_recon = (float*)d_out + (size_t)N * G;

    const int NG = N * G;
    const int ngBlocks = (NG + 255) / 256;
    const dim3 attnGrid(N / 128, N / 64);
    const dim3 blk(256);

    // ---- weight transposes (f32 -> bf16, [K,Nc] -> [Nc,K]) ----
    {
        int nb = (D * D + 255) / 256;
        for (int i = 0; i < 5; ++i)
            tc_kernel<<<nb, blk, 0, stream>>>(enc_W[i], encWt[i], D, D);
        for (int i = 0; i < 5; ++i)
            tc_kernel<<<nb, blk, 0, stream>>>(dec_W[i], decWt[i], D, D);
        tc_kernel<<<nb, blk, 0, stream>>>(head_W, headWt, D, D);
        int nb1 = (D * F + 255) / 256;
        tc_kernel<<<nb1, blk, 0, stream>>>(ff_W1, W1t, D, F);
        tc_kernel<<<nb1, blk, 0, stream>>>(ff_W2, W2t, F, D);
    }

    // ---- drop_x and x_init ----
    convert_kernel<<<(N * D + 255) / 256, blk, 0, stream>>>(x, dropf, dropb, N * D);
    zero_drop_kernel<<<ngBlocks, blk, 0, stream>>>(drop_idx, dropf, dropb, NG);
    gather_kernel<<<ngBlocks, blk, 0, stream>>>(x, drop_idx, out_init, NG);

    // EPI bits: 1=bias 2=gelu 4=f32 8=bf16 16=bf16^T
    // ---- encoder projections ----
    {
        dim3 g(N / 128, D / 128);
        gemm_bf16_kernel<8><<<g, blk, 0, stream>>>(dropb, encWt[0], nullptr, nullptr, qr_b, nullptr, N, D, D);
        gemm_bf16_kernel<8><<<g, blk, 0, stream>>>(dropb, encWt[1], nullptr, nullptr, kr_b, nullptr, N, D, D);
        gemm_bf16_kernel<8><<<g, blk, 0, stream>>>(dropb, encWt[2], nullptr, nullptr, qf_b, nullptr, N, D, D);
        gemm_bf16_kernel<8><<<g, blk, 0, stream>>>(dropb, encWt[3], nullptr, nullptr, kf_b, nullptr, N, D, D);
        gemm_bf16_kernel<16><<<g, blk, 0, stream>>>(dropb, encWt[4], nullptr, nullptr, nullptr, vt_b, N, D, D);
    }

    // ---- encoder attention ----
    hipMemsetAsync(rowmax_u, 0, (size_t)N * 4, stream);
    hipMemsetAsync(rowsum, 0, (size_t)N * 4, stream);
    attn_scores_kernel<0><<<attnGrid, blk, 0, stream>>>(qr_b, kr_b, qf_b, kf_b, rmask, fmask,
                                                        rowmax_u, rowsum, Pmat);
    attn_scores_kernel<1><<<attnGrid, blk, 0, stream>>>(qr_b, kr_b, qf_b, kf_b, rmask, fmask,
                                                        rowmax_u, rowsum, Pmat);
    {
        dim3 g(N / 128, D / 128);
        gemm_bf16_kernel<4><<<g, blk, 0, stream>>>(Pmat, vt_b, nullptr, pv_f, nullptr, nullptr, N, D, N);
    }
    ln_kernel<<<N / 8, blk, 0, stream>>>(dropf, pv_f, rowsum, ln1_g, ln1_b, hid_f, hid_b);

    // ---- FFN ----
    {
        dim3 g1(N / 128, F / 128);
        gemm_bf16_kernel<1 | 2 | 8><<<g1, blk, 0, stream>>>(hid_b, W1t, ff_b1, nullptr, ff1_b, nullptr, N, F, D);
        dim3 g2(N / 128, D / 128);
        gemm_bf16_kernel<1 | 4><<<g2, blk, 0, stream>>>(ff1_b, W2t, ff_b2, pv_f, nullptr, nullptr, N, D, F);
    }
    ln_kernel<<<N / 8, blk, 0, stream>>>(hid_f, pv_f, nullptr, ln2_g, ln2_b, hid2_f, hid2_b);

    // ---- decoder projections ----
    {
        dim3 g(N / 128, D / 128);
        gemm_bf16_kernel<8><<<g, blk, 0, stream>>>(hid2_b, decWt[0], nullptr, nullptr, qr_b, nullptr, N, D, D);
        gemm_bf16_kernel<8><<<g, blk, 0, stream>>>(hid2_b, decWt[1], nullptr, nullptr, kr_b, nullptr, N, D, D);
        gemm_bf16_kernel<8><<<g, blk, 0, stream>>>(hid2_b, decWt[2], nullptr, nullptr, qf_b, nullptr, N, D, D);
        gemm_bf16_kernel<8><<<g, blk, 0, stream>>>(hid2_b, decWt[3], nullptr, nullptr, kf_b, nullptr, N, D, D);
        gemm_bf16_kernel<16><<<g, blk, 0, stream>>>(hid2_b, decWt[4], nullptr, nullptr, nullptr, vt_b, N, D, D);
    }

    // ---- decoder attention ----
    hipMemsetAsync(rowmax_u, 0, (size_t)N * 4, stream);
    hipMemsetAsync(rowsum, 0, (size_t)N * 4, stream);
    attn_scores_kernel<0><<<attnGrid, blk, 0, stream>>>(qr_b, kr_b, qf_b, kf_b, rmask, fmask,
                                                        rowmax_u, rowsum, Pmat);
    attn_scores_kernel<1><<<attnGrid, blk, 0, stream>>>(qr_b, kr_b, qf_b, kf_b, rmask, fmask,
                                                        rowmax_u, rowsum, Pmat);
    {
        dim3 g(N / 128, D / 128);
        gemm_bf16_kernel<4><<<g, blk, 0, stream>>>(Pmat, vt_b, nullptr, pv_f, nullptr, nullptr, N, D, N);
    }
    ln_kernel<<<N / 8, blk, 0, stream>>>(hid2_f, pv_f, rowsum, ln3_g, ln3_b, nullptr, recon_b);

    // ---- head + x_recon ----
    {
        dim3 g(N / 128, D / 128);
        gemm_bf16_kernel<1 | 4><<<g, blk, 0, stream>>>(recon_b, headWt, head_b, head_f, nullptr, nullptr, N, D, D);
    }
    gather_kernel<<<ngBlocks, blk, 0, stream>>>(head_f, drop_idx, out_recon, NG);
}
